// QASPTransformer_86981677679383
// MI455X (gfx1250) — matrix operand order repaired and checked by reading, not compiled, not run on hardware
//
#include <hip/hip_runtime.h>
#include <hip/hip_bf16.h>

typedef __attribute__((ext_vector_type(16))) _Float16 v16h;
typedef __attribute__((ext_vector_type(8)))  float    v8f;

constexpr int CB = 2, CT = 1024, CD = 1024, CH = 16, CDH = 64;
constexpr int CFF = 4096, CL = 2, CV = 32000;
constexpr int CMB = 32, CBS = 32, CTOPK = 8, CKEEP = 128;
constexpr float CEPS = 1e-6f;
constexpr float CTHETA = 10000.f;
constexpr float CSCALE = 0.125f;      // 1/sqrt(DH=64)
constexpr float CPI = 3.14159265358979323846f;

// ---------------- CDNA5 async copy (ASYNCcnt-tracked, VMEM -> LDS direct) ----------------

#define USE_ASYNC_LDS 1

#if USE_ASYNC_LDS
__device__ __forceinline__ void async_ld_b128(unsigned lds_off, const void* g) {
  // GLOBAL_LOAD_ASYNC_TO_LDS_B128: VDST = per-lane LDS byte offset, VADDR = 64b global addr
  asm volatile("global_load_async_to_lds_b128 %0, %1, off"
               :: "v"(lds_off), "v"((unsigned long long)(uintptr_t)g)
               : "memory");
}
__device__ __forceinline__ void wait_async0() {
  asm volatile("s_wait_asynccnt 0x0" ::: "memory");
}
#endif

// ---------------- WMMA helpers (CDNA5 16x16x32 f16 -> f32) ----------------

__device__ __forceinline__ v8f wmma_f16(v16h a, v16h b, v8f c) {
  // 8 args: (neg_a, A, neg_b, B, c_mod, C, reuse_a, reuse_b)
  return __builtin_amdgcn_wmma_f32_16x16x32_f16(false, a, false, b, (short)0, c, false, false);
}

// A fragment (16x32, row-per-lane).  Lanes 0-15: K 0..15; lanes 16-31: K 16..31.
__device__ __forceinline__ v16h afrag_f32(const float* base, int hi) {
  v16h r;
  const float* p0 = base + hi * 16;
  const float* p1 = base + hi * 16 + 8;
#pragma unroll
  for (int i = 0; i < 8; i++) { r[i] = (_Float16)p0[i]; r[i + 8] = (_Float16)p1[i]; }
  return r;
}
// B fragment (32x16, column-per-lane). Lanes 0-15: K 0..15; lanes 16-31: K 16..31.
__device__ __forceinline__ v16h bfrag_f32(const float* base, int hi) {
  v16h r;
  const float* p = base + hi * 16;
#pragma unroll
  for (int i = 0; i < 16; i++) r[i] = (_Float16)p[i];
  return r;
}
__device__ __forceinline__ v16h bfrag_h(const _Float16* base, int hi) {
  v16h r;
  const _Float16* p = base + hi * 16;
#pragma unroll
  for (int i = 0; i < 16; i++) r[i] = p[i];
  return r;
}

// ---------------- Generic WMMA GEMM:  C[z] = A[z] (MxK) * W[z] (KxN) (+R) ----------------
// f32 in/out, f16 WMMA with f32 accumulate. Requires M%64==0, N%128==0, K%32==0.
// Block tile 64x128x32; 256 threads = 8 waves arranged 2(M) x 4(N); wave tile 32x32
// -> 4 WMMA per K-step per wave.  A tile staged f32 via global_load_async_to_lds_b128;
// B tile transposed+converted to f16 with packed b32 LDS stores.

__global__ __launch_bounds__(256) void k_gemm(
    const float* __restrict__ A, const float* __restrict__ W,
    const float* __restrict__ R, float* __restrict__ C,
    int Mdim, int Ndim, int Kdim, long long sA, long long sB, long long sC) {
  A += (long long)blockIdx.z * sA;
  W += (long long)blockIdx.z * sB;
  C += (long long)blockIdx.z * sC;
  if (R) R += (long long)blockIdx.z * sC;

  constexpr int LDAF = 36;                        // A tile row stride (f32) : 144B, 16B aligned
  constexpr int LDB  = 40;                        // B tile row stride (f16) : +8 pad
  __shared__ float    lsa32[64 * LDAF];           // A tile 64x32 raw f32 (async target)
  __shared__ _Float16 lsb[128 * LDB];             // B tile transposed [n][k] f16

  const int tid = threadIdx.x;
  const int lane = tid & 31;
  const int wave = tid >> 5;
  const int lane16 = lane & 15, hi = (lane >> 4) & 1;
  const int wm = wave >> 2, wn = wave & 3;        // 2 x 4 wave grid
  const int m0 = blockIdx.y * 64, n0 = blockIdx.x * 128;

  const int bn8 = (tid & 15) * 8;                 // B loader: n in [0,128) by 8
  const int bk2 = (tid >> 4) * 2;                 // B loader: even k in [0,32)
#if USE_ASYNC_LDS
  const unsigned lsa_base = (unsigned)(uintptr_t)&lsa32[0];  // LDS aperture: low 32b = LDS offset
#else
  const int ar = tid >> 2, ac = (tid & 3) * 8;
#endif

  v8f a00 = {}, a01 = {}, a10 = {}, a11 = {};

  for (int k0 = 0; k0 < Kdim; k0 += 32) {
#if USE_ASYNC_LDS
    // ---- A tile: 512 x b128 segments, 2 per thread, direct to LDS (ASYNCcnt)
#pragma unroll
    for (int s = 0; s < 2; s++) {
      int seg = tid * 2 + s;
      int row = seg >> 3, c4 = (seg & 7) * 4;
      const float* gp = A + (long long)(m0 + row) * Kdim + k0 + c4;
      async_ld_b128(lsa_base + (unsigned)((row * LDAF + c4) * 4), gp);
    }
#else
    const float* ap = A + (long long)(m0 + ar) * Kdim + k0 + ac;
#pragma unroll
    for (int i = 0; i < 8; i++) lsa32[ar * LDAF + ac + i] = ap[i];
#endif
    // ---- B tile: transpose+convert, pack 2 k-adjacent halves per b32 store
    const float* bp0 = W + (long long)(k0 + bk2) * Ndim + n0 + bn8;
    const float* bp1 = bp0 + Ndim;
#pragma unroll
    for (int i = 0; i < 8; i++) {
      union { _Float16 h[2]; unsigned u; } t;
      t.h[0] = (_Float16)bp0[i];
      t.h[1] = (_Float16)bp1[i];
      *(unsigned*)&lsb[(bn8 + i) * LDB + bk2] = t.u;
    }
    if (k0 + 32 < Kdim) {                         // global_prefetch_b8 next K tile
      __builtin_prefetch(W + (long long)(k0 + 32 + bk2) * Ndim + n0 + bn8, 0, 0);
      __builtin_prefetch(A + (long long)(m0 + (tid >> 2)) * Kdim + k0 + 32 + (tid & 3) * 8, 0, 0);
    }
#if USE_ASYNC_LDS
    wait_async0();                                // our asyncs landed in LDS
#endif
    __syncthreads();

    v16h af0 = afrag_f32(&lsa32[(wm * 32 + lane16) * LDAF], hi);
    v16h af1 = afrag_f32(&lsa32[(wm * 32 + 16 + lane16) * LDAF], hi);
    v16h bf0 = bfrag_h(&lsb[(wn * 32 + lane16) * LDB], hi);
    v16h bf1 = bfrag_h(&lsb[(wn * 32 + 16 + lane16) * LDB], hi);
    a00 = wmma_f16(af0, bf0, a00);
    a01 = wmma_f16(af0, bf1, a01);
    a10 = wmma_f16(af1, bf0, a10);
    a11 = wmma_f16(af1, bf1, a11);
    __syncthreads();
  }

  const int rbase = m0 + wm * 32;
  const int cbase = n0 + wn * 32;
#pragma unroll
  for (int j = 0; j < 8; j++) {                   // C layout: VGPR j <-> rows j / j+8
    long long r0 = (long long)(rbase + j + 8 * hi) * Ndim;
    long long r1 = (long long)(rbase + 16 + j + 8 * hi) * Ndim;
    int c0 = cbase + lane16, c1 = c0 + 16;
    float v00 = a00[j], v01 = a01[j], v10 = a10[j], v11 = a11[j];
    if (R) { v00 += R[r0 + c0]; v01 += R[r0 + c1]; v10 += R[r1 + c0]; v11 += R[r1 + c1]; }
    C[r0 + c0] = v00; C[r0 + c1] = v01;
    C[r1 + c0] = v10; C[r1 + c1] = v11;
  }
}

// ---------------- Block-sparse attention (<=9 KV blocks per query block) ----------------

constexpr int NKMAX = 9 * 32;

__global__ __launch_bounds__(128) void k_attn(
    const float* __restrict__ q, const float* __restrict__ k,
    const float* __restrict__ v, const int* __restrict__ sel,
    float* __restrict__ o) {
  const int qb = blockIdx.x, h = blockIdx.y, b = blockIdx.z;
  const int tid = threadIdx.x;
  const int lane = tid & 31, wave = tid >> 5;
  const int lane16 = lane & 15, hi = (lane >> 4) & 1;

  __shared__ int kvblk[9];
  __shared__ int nblk_s;
  __shared__ float ss[32][NKMAX + 8];
  __shared__ float redm[32][4], redsum[32][4];

  if (tid == 0) {
    int n = 0;
    for (int kb = 0; kb <= qb; kb++)
      if (kb == qb || sel[b * CMB + kb]) { if (n < 9) kvblk[n++] = kb; }
    nblk_s = n;
  }
  __syncthreads();
  const int nblk = nblk_s;
  const int nk = nblk * 32;

  // ---- S = scale * Q K^T   (q/k rows contiguous over DH -> fragments straight from global)
  for (int tile = wave; tile < 4 * nblk; tile += 4) {
    int qh = tile & 1, kt = tile >> 1;
    int tq16 = qb * 32 + qh * 16 + lane16;
    const float* qp = q + ((long long)(b * CT + tq16) * CD + h * CDH);
    int jb = kvblk[kt >> 1];
    int tk0 = jb * 32 + (kt & 1) * 16;
    const float* kp = k + ((long long)(b * CT + tk0 + lane16) * CD + h * CDH);
    v8f acc = {};
#pragma unroll
    for (int c = 0; c < 2; c++) {                // DH=64 -> two K=32 WMMA steps
      v16h af = afrag_f32(qp + 32 * c, hi);
      v16h bf = bfrag_f32(kp + 32 * c, hi);
      acc = wmma_f16(af, bf, acc);
    }
#pragma unroll
    for (int j = 0; j < 8; j++) {
      int rloc = qh * 16 + j + 8 * hi;
      int tq = qb * 32 + rloc;
      int tk = tk0 + lane16;
      float sv = acc[j] * CSCALE;
      if (tk > tq) sv = -1e9f;                   // causal mask (only bites in diagonal block)
      ss[rloc][kt * 16 + lane16] = sv;
    }
  }
  __syncthreads();

  // ---- row softmax over the compacted key list (4 threads per row)
  {
    int row = tid >> 2, sub = tid & 3;
    float mx = -1e30f;
    for (int c2 = sub; c2 < nk; c2 += 4) mx = fmaxf(mx, ss[row][c2]);
    redm[row][sub] = mx;
    __syncthreads();
    float m = fmaxf(fmaxf(redm[row][0], redm[row][1]),
                    fmaxf(redm[row][2], redm[row][3]));
    float sum = 0.f;
    for (int c2 = sub; c2 < nk; c2 += 4) {
      float e = __expf(ss[row][c2] - m);
      ss[row][c2] = e;
      sum += e;
    }
    redsum[row][sub] = sum;
    __syncthreads();
    float inv = 1.f / (redsum[row][0] + redsum[row][1] + redsum[row][2] + redsum[row][3]);
    for (int c2 = sub; c2 < nk; c2 += 4) ss[row][c2] *= inv;
  }
  __syncthreads();

  // ---- O = P V  (8 output tiles of 16x16; contraction over nk keys in chunks of 32)
  for (int tile = wave; tile < 8; tile += 4) {
    int qh = tile & 1, dt = tile >> 1;
    v8f acc = {};
    for (int kk = 0; kk < nk; kk += 32) {
      v16h af = afrag_f32(&ss[qh * 16 + lane16][kk], hi);
      int jb = kvblk[kk >> 5];
      const float* vp = v + ((long long)(b * CT + jb * 32 + hi * 16) * CD + h * CDH + dt * 16 + lane16);
      v16h bf;
#pragma unroll
      for (int i = 0; i < 16; i++) bf[i] = (_Float16)vp[(long long)i * CD];
      acc = wmma_f16(af, bf, acc);
    }
#pragma unroll
    for (int j = 0; j < 8; j++) {
      int rloc = qh * 16 + j + 8 * hi;
      int tq = qb * 32 + rloc;
      o[(long long)(b * CT + tq) * CD + h * CDH + dt * 16 + lane16] = acc[j];
    }
  }
}

// ---------------- Small kernels ----------------

__global__ void k_embed(const int* __restrict__ ids, const float* __restrict__ emb,
                        float* __restrict__ x) {
  int bt = blockIdx.x;
  int id = ids[bt];
  const float4* s = (const float4*)(emb + (long long)id * CD);
  float4* d = (float4*)(x + (long long)bt * CD);
  d[threadIdx.x] = s[threadIdx.x];
}

// Closed-form low-pass projector: irfft(keep K bins) == circulant Dirichlet kernel.
__global__ void k_dirichlet(float* __restrict__ P) {
  int idx = blockIdx.x * blockDim.x + threadIdx.x;
  if (idx >= CT * CT) return;
  int t = idx / CT, s2 = idx % CT;
  int d = t - s2;
  float val;
  if (d == 0) val = (float)(2 * CKEEP - 1);
  else {
    float xx = CPI * (float)d / (float)CT;
    val = sinf((float)(2 * CKEEP - 1) * xx) / sinf(xx);
  }
  P[idx] = val / (float)CT;
}

__global__ __launch_bounds__(256) void k_rho(const float* __restrict__ hlp,
                                             const float* __restrict__ x,
                                             float* __restrict__ rho) {
  int bt = blockIdx.x, tid = threadIdx.x;
  __shared__ float sa[256], sb[256];
  float a = 0.f, c = 0.f;
  for (int i = tid; i < CD; i += 256) {
    float hv = hlp[(long long)bt * CD + i]; a += hv * hv;
    float xv = x[(long long)bt * CD + i];   c += xv * xv;
  }
  sa[tid] = a; sb[tid] = c;
  __syncthreads();
  for (int s = 128; s > 0; s >>= 1) {
    if (tid < s) { sa[tid] += sa[tid + s]; sb[tid] += sb[tid + s]; }
    __syncthreads();
  }
  if (tid == 0) rho[bt] = sa[0] / (sb[0] + CEPS);
}

__global__ void k_topk(const float* __restrict__ rho, int* __restrict__ sel) {
  int b = blockIdx.x, m = threadIdx.x;   // 32 threads = one lane per block
  __shared__ float bq[CMB];
  __shared__ int ch[CMB];
  float s = 0.f;
  for (int i = 0; i < CBS; i++) s += rho[(long long)b * CT + m * CBS + i];
  bq[m] = s; ch[m] = 0;
  __syncthreads();
  if (m == 0) {
    for (int it = 0; it < CTOPK; it++) {
      float best = -1e30f; int bi = 0;
      for (int j = 0; j < CMB; j++)
        if (!ch[j] && bq[j] > best) { best = bq[j]; bi = j; }
      ch[bi] = 1;
    }
  }
  __syncthreads();
  sel[b * CMB + m] = ch[m];
}

__global__ __launch_bounds__(256) void k_rms(const float* __restrict__ x,
                                             const float* __restrict__ w,
                                             float* __restrict__ out) {
  int bt = blockIdx.x, tid = threadIdx.x;
  __shared__ float sa[256];
  float a = 0.f;
  for (int i = tid; i < CD; i += 256) { float v = x[(long long)bt * CD + i]; a += v * v; }
  sa[tid] = a;
  __syncthreads();
  for (int s = 128; s > 0; s >>= 1) {
    if (tid < s) sa[tid] += sa[tid + s];
    __syncthreads();
  }
  float nrm = rsqrtf(sa[0] / (float)CD + CEPS);
  for (int i = tid; i < CD; i += 256)
    out[(long long)bt * CD + i] = x[(long long)bt * CD + i] * nrm * w[i];
}

__global__ void k_rope(float* __restrict__ qk) {
  int idx = blockIdx.x * blockDim.x + threadIdx.x;   // over B*T*H*32 rotation pairs
  int hd = idx & 31;
  int r = idx >> 5;
  int hh = r % CH; r /= CH;
  int t = r % CT;
  int b = r / CT;
  float inv = powf(CTHETA, -(float)hd / 32.f);
  float ang = (float)t * inv;
  float cc = cosf(ang), sn = sinf(ang);
  long long base = (long long)(b * CT + t) * CD + hh * CDH;
  float x0 = qk[base + hd], x1 = qk[base + hd + 32];
  qk[base + hd]      = x0 * cc - x1 * sn;
  qk[base + hd + 32] = x1 * cc + x0 * sn;
}

__global__ void k_silu_mul(float* __restrict__ f1, const float* __restrict__ f3, long long n) {
  long long i = (long long)blockIdx.x * blockDim.x + threadIdx.x;
  if (i < n) { float a = f1[i]; f1[i] = a / (1.f + __expf(-a)) * f3[i]; }
}

// ---------------- Host orchestration ----------------

extern "C" void kernel_launch(void* const* d_in, const int* in_sizes, int n_in,
                              void* d_out, int out_size, void* d_ws, size_t ws_size,
                              hipStream_t stream) {
  (void)in_sizes; (void)n_in; (void)out_size; (void)ws_size;
  const int*   ids        = (const int*)d_in[0];
  const float* emb        = (const float*)d_in[1];
  const float* wq         = (const float*)d_in[2];
  const float* wk         = (const float*)d_in[3];
  const float* wv         = (const float*)d_in[4];
  const float* wo         = (const float*)d_in[5];
  const float* w1         = (const float*)d_in[6];
  const float* w2         = (const float*)d_in[7];
  const float* w3         = (const float*)d_in[8];
  const float* attn_norm  = (const float*)d_in[9];
  const float* ffn_norm   = (const float*)d_in[10];
  const float* final_norm = (const float*)d_in[11];
  const float* lm_head    = (const float*)d_in[12];
  float* out = (float*)d_out;

  const long long BTD = (long long)CB * CT * CD;
  const long long BTF = (long long)CB * CT * CFF;
  float* ws  = (float*)d_ws;
  float* x   = ws;
  float* hb  = x   + BTD;
  float* qb_ = hb  + BTD;
  float* kb_ = qb_ + BTD;
  float* vb_ = kb_ + BTD;
  float* ob_ = vb_ + BTD;
  float* f1  = ob_ + BTD;
  float* f3  = f1  + BTF;
  float* Pm  = f3  + BTF;
  float* hlp = Pm  + (long long)CT * CT;
  float* rho = hlp + BTD;
  int*   sel = (int*)(rho + (long long)CB * CT);

  const int BT = CB * CT;

  k_embed<<<BT, 256, 0, stream>>>(ids, emb, x);
  k_dirichlet<<<(CT * CT) / 256, 256, 0, stream>>>(Pm);
  // h_lp[b] = P @ x[b]   (batched over z; A shared)
  k_gemm<<<dim3(CD / 128, CT / 64, CB), 256, 0, stream>>>(
      Pm, x, nullptr, hlp, CT, CD, CT, 0, (long long)CT * CD, (long long)CT * CD);
  k_rho<<<BT, 256, 0, stream>>>(hlp, x, rho);
  k_topk<<<CB, 32, 0, stream>>>(rho, sel);

  const dim3 gdd(CD / 128, BT / 64, 1);
  const dim3 gdf(CFF / 128, BT / 64, 1);
  const int ropeN = CB * CT * CH * 32;

  for (int l = 0; l < CL; l++) {
    const float* wq_l = wq + (long long)l * CD * CD;
    const float* wk_l = wk + (long long)l * CD * CD;
    const float* wv_l = wv + (long long)l * CD * CD;
    const float* wo_l = wo + (long long)l * CD * CD;
    const float* w1_l = w1 + (long long)l * CD * CFF;
    const float* w2_l = w2 + (long long)l * CFF * CD;
    const float* w3_l = w3 + (long long)l * CD * CFF;

    k_rms<<<BT, 256, 0, stream>>>(x, attn_norm + l * CD, hb);
    k_gemm<<<gdd, 256, 0, stream>>>(hb, wq_l, nullptr, qb_, BT, CD, CD, 0, 0, 0);
    k_gemm<<<gdd, 256, 0, stream>>>(hb, wk_l, nullptr, kb_, BT, CD, CD, 0, 0, 0);
    k_gemm<<<gdd, 256, 0, stream>>>(hb, wv_l, nullptr, vb_, BT, CD, CD, 0, 0, 0);
    k_rope<<<ropeN / 256, 256, 0, stream>>>(qb_);
    k_rope<<<ropeN / 256, 256, 0, stream>>>(kb_);
    k_attn<<<dim3(CT / 32, CH, CB), 128, 0, stream>>>(qb_, kb_, vb_, sel, ob_);
    k_gemm<<<gdd, 256, 0, stream>>>(ob_, wo_l, x, x, BT, CD, CD, 0, 0, 0);   // + residual

    k_rms<<<BT, 256, 0, stream>>>(x, ffn_norm + l * CD, hb);
    k_gemm<<<gdf, 256, 0, stream>>>(hb, w1_l, nullptr, f1, BT, CFF, CD, 0, 0, 0);
    k_gemm<<<gdf, 256, 0, stream>>>(hb, w3_l, nullptr, f3, BT, CFF, CD, 0, 0, 0);
    k_silu_mul<<<(int)(BTF / 256), 256, 0, stream>>>(f1, f3, BTF);
    k_gemm<<<gdd, 256, 0, stream>>>(f1, w2_l, x, x, BT, CD, CFF, 0, 0, 0);   // + residual
  }

  k_rms<<<BT, 256, 0, stream>>>(x, final_norm, hb);
  k_gemm<<<dim3(CV / 128, BT / 64, 1), 256, 0, stream>>>(
      hb, lm_head, nullptr, out, BT, CV, CD, 0, 0, 0);
}
